// InternAttention_22041772163585
// MI455X (gfx1250) — matrix operand order repaired and checked by reading, not compiled, not run on hardware
//
#include <hip/hip_runtime.h>

#define DEV __device__ __forceinline__

typedef __attribute__((ext_vector_type(16))) _Float16 v16h;
typedef __attribute__((ext_vector_type(8)))  _Float16 v8h;
typedef __attribute__((ext_vector_type(4)))  _Float16 v4h;
typedef __attribute__((ext_vector_type(8)))  float    v8f;

constexpr int B_ = 4, S_ = 2048, E_ = 1024, H_ = 16, D_ = 64;

DEV v8f wmma_f16(v16h a, v16h b, v8f c) {
  return __builtin_amdgcn_wmma_f32_16x16x32_f16(false, a, false, b, (short)0, c,
                                                false, false);
}

DEV v8f zero8() {
  v8f z;
#pragma unroll
  for (int i = 0; i < 8; ++i) z[i] = 0.f;
  return z;
}

DEV unsigned lds_off(const void* p) {  // addrspace(3)->flat keeps LDS offset in low 32 bits
  return (unsigned)(uintptr_t)p;
}

// CDNA5 async DMA: global -> LDS, 16B per lane, tracked by ASYNCcnt.
DEV void async_ld_b128(unsigned lds_byte, unsigned gvoff, uint64_t sbase) {
  asm volatile("global_load_async_to_lds_b128 %0, %1, %2"
               :: "v"(lds_byte), "v"(gvoff), "s"(sbase) : "memory");
}
#define WAIT_ASYNC(N) asm volatile("s_wait_asynccnt %0" :: "n"(N) : "memory")
#define WAIT_DS0()    asm volatile("s_wait_dscnt 0" ::: "memory")

// ---- fragment loaders (gfx1250 wave32 WMMA VGPR layouts) -------------------
DEV v16h load_A_frag(const _Float16* p, int ld, int lane) {
  const int half = lane >> 4, m = lane & 15;
  v16h a;
#pragma unroll
  for (int v = 0; v < 8; ++v) {
    const int k = 2 * v + 16 * half;
    a[2 * v]     = p[m * ld + k];
    a[2 * v + 1] = p[m * ld + k + 1];
  }
  return a;
}

DEV v16h load_A_frag_scaled(const _Float16* p, int ld, int lane, float scale) {
  const int half = lane >> 4, m = lane & 15;
  v16h a;
#pragma unroll
  for (int v = 0; v < 8; ++v) {
    const int k = 2 * v + 16 * half;
    a[2 * v]     = (_Float16)(scale * (float)p[m * ld + k]);
    a[2 * v + 1] = (_Float16)(scale * (float)p[m * ld + k + 1]);
  }
  return a;
}

// B 32x16 (KxN) from tile stored [N][ld] (n-major, k contiguous)
DEV v16h load_B_frag_nk(const _Float16* p, int ld, int lane) {
  const int half = lane >> 4, n = lane & 15;
  v16h b;
#pragma unroll
  for (int v = 0; v < 8; ++v) {
    const int k = 2 * v + 16 * half;
    b[2 * v]     = p[n * ld + k];
    b[2 * v + 1] = p[n * ld + k + 1];
  }
  return b;
}

// B 32x16 (KxN) from tile stored [K][ld] (k-major, n contiguous)
DEV v16h load_B_frag_kn(const _Float16* p, int ld, int lane) {
  const int half = lane >> 4, n = lane & 15;
  v16h b;
#pragma unroll
  for (int v = 0; v < 8; ++v) {
    const int k = 2 * v + 16 * half;
    b[2 * v]     = p[k * ld + n];
    b[2 * v + 1] = p[(k + 1) * ld + n];
  }
  return b;
}

// ---- f32 -> f16 conversion pass (enables pure async-DMA staging) -----------
__global__ __launch_bounds__(256) void cvt_f32_f16(const float* __restrict__ s,
                                                   _Float16* __restrict__ d,
                                                   int n) {
  const int i = (blockIdx.x * 256 + threadIdx.x) * 4;
  if (i < n) {
    const float4 f = *(const float4*)(s + i);
    v4h h;
    h[0] = (_Float16)f.x; h[1] = (_Float16)f.y;
    h[2] = (_Float16)f.z; h[3] = (_Float16)f.w;
    *(v4h*)(d + i) = h;
  }
}

// ---- GEMM: Out[m][n] = sum_k A[m][k]*W[n][k] + bias[n] ---------------------
// Block tile 256(M) x 64(N), BK=32; 8 waves each computing 32x64 (8 WMMAs per
// k-step from 6 fragment loads). Double-buffered async-DMA staging.
template <typename TO>
__global__ __launch_bounds__(256) void gemm_f16_bias(
    const _Float16* __restrict__ A, const _Float16* __restrict__ W,
    const float* __restrict__ bias, TO* __restrict__ Out, int M, int N, int K) {
  __shared__ _Float16 sA[2][256 * 32];
  __shared__ _Float16 sB[2][64 * 32];
  const int tid = threadIdx.x, lane = tid & 31, wave = tid >> 5;
  const int m0 = blockIdx.y * 256, n0 = blockIdx.x * 64;

  // per-lane async geometry: A tile 256x32 = 4 x b128/lane, B tile 64x32 = 1
  const int eA = tid * 8, rowA = eA >> 5, colA = eA & 31;  // rowA in [0,64)
  unsigned voffA[4];
#pragma unroll
  for (int p = 0; p < 4; ++p)
    voffA[p] = (unsigned)(((rowA + p * 64) * K + colA) * 2);
  const unsigned voffB = voffA[0];
  const unsigned ldsA[2] = {lds_off(&sA[0][0]), lds_off(&sA[1][0])};
  const unsigned ldsB[2] = {lds_off(&sB[0][0]), lds_off(&sB[1][0])};

  auto stage = [&](int kk, int buf) {
    const uint64_t sbA = (uint64_t)(uintptr_t)(A + (size_t)m0 * K + kk);
    const uint64_t sbB = (uint64_t)(uintptr_t)(W + (size_t)n0 * K + kk);
#pragma unroll
    for (int p = 0; p < 4; ++p)
      async_ld_b128(ldsA[buf] + tid * 16 + p * 4096, voffA[p], sbA);
    async_ld_b128(ldsB[buf] + tid * 16, voffB, sbB);
  };

  v8f acc[2][4];
#pragma unroll
  for (int mi = 0; mi < 2; ++mi)
#pragma unroll
    for (int t = 0; t < 4; ++t) acc[mi][t] = zero8();

  const int NK = K / 32;
  stage(0, 0);
  for (int ki = 0; ki < NK; ++ki) {
    if (ki + 1 < NK) { stage((ki + 1) * 32, (ki + 1) & 1); WAIT_ASYNC(5); }
    else             { WAIT_ASYNC(0); }
    __syncthreads();
    const _Float16* cA = sA[ki & 1];
    const _Float16* cB = sB[ki & 1];
    const v16h aF0 = load_A_frag(&cA[(wave * 32) * 32], 32, lane);
    const v16h aF1 = load_A_frag(&cA[(wave * 32 + 16) * 32], 32, lane);
#pragma unroll
    for (int t = 0; t < 4; ++t) {
      const v16h bF = load_B_frag_nk(&cB[(t * 16) * 32], 32, lane);
      acc[0][t] = wmma_f16(aF0, bF, acc[0][t]);
      acc[1][t] = wmma_f16(aF1, bF, acc[1][t]);
    }
    __syncthreads();
  }

  const int nn = lane & 15, half = lane >> 4;
#pragma unroll
  for (int mi = 0; mi < 2; ++mi) {
#pragma unroll
    for (int t = 0; t < 4; ++t) {
      const int n = n0 + t * 16 + nn;
      const float bv = bias[n];
#pragma unroll
      for (int v = 0; v < 8; ++v) {
        const int m = m0 + wave * 32 + mi * 16 + v + 8 * half;
        Out[(size_t)m * N + n] = (TO)(acc[mi][t][v] + bv);
      }
    }
  }
}

// ---- Flash attention: block = (b, h, 128 q rows); wave owns 16 rows --------
// 64-key chunks, double-buffered async K/V staging, online softmax.
__global__ __launch_bounds__(256) void flash_attn_f16(
    const _Float16* __restrict__ qkv, _Float16* __restrict__ ctx) {
  __shared__ _Float16 sK[2][64 * 64];
  __shared__ _Float16 sV[2][64 * 64];
  __shared__ _Float16 sP[8][16 * 64];
  const int tid = threadIdx.x, lane = tid & 31, wave = tid >> 5;
  const int half = lane >> 4, nn = lane & 15;
  const int b = blockIdx.z, h = blockIdx.y;
  const int q0 = blockIdx.x * 128 + wave * 16;

  // Q fragments, 1/sqrt(D) folded in (exact power of two in f16)
  const _Float16* qp = qkv + ((size_t)(b * S_ + q0) * (3 * E_)) + h * D_;
  const v16h qA0 = load_A_frag_scaled(qp, 3 * E_, lane, 0.125f);
  const v16h qA1 = load_A_frag_scaled(qp + 32, 3 * E_, lane, 0.125f);

  // per-lane async geometry for 64x64 K/V tiles: 2 x b128 per lane per tile
  const int e0 = tid * 8, rowT = e0 >> 6, colT = e0 & 63;
  const unsigned vK0 = (unsigned)(((rowT)      * 3 * E_ + E_     + colT) * 2);
  const unsigned vK1 = (unsigned)(((rowT + 32) * 3 * E_ + E_     + colT) * 2);
  const unsigned vV0 = (unsigned)(((rowT)      * 3 * E_ + 2 * E_ + colT) * 2);
  const unsigned vV1 = (unsigned)(((rowT + 32) * 3 * E_ + 2 * E_ + colT) * 2);
  const unsigned ldsK[2] = {lds_off(&sK[0][0]), lds_off(&sK[1][0])};
  const unsigned ldsV[2] = {lds_off(&sV[0][0]), lds_off(&sV[1][0])};

  auto stage = [&](int kc, int buf) {
    const uint64_t sb =
        (uint64_t)(uintptr_t)(qkv + (size_t)(b * S_ + kc) * (3 * E_) + h * D_);
    async_ld_b128(ldsK[buf] + tid * 16,        vK0, sb);
    async_ld_b128(ldsK[buf] + tid * 16 + 4096, vK1, sb);
    async_ld_b128(ldsV[buf] + tid * 16,        vV0, sb);
    async_ld_b128(ldsV[buf] + tid * 16 + 4096, vV1, sb);
  };

  v8f O[4];
#pragma unroll
  for (int t = 0; t < 4; ++t) O[t] = zero8();
  float rmax[8], rsum[8];
#pragma unroll
  for (int v = 0; v < 8; ++v) { rmax[v] = -3.0e38f; rsum[v] = 0.f; }
  _Float16* myP = &sP[wave][0];

  const int NC = S_ / 64;
  stage(0, 0);
  for (int c = 0; c < NC; ++c) {
    if (c + 1 < NC) { stage((c + 1) * 64, (c + 1) & 1); WAIT_ASYNC(4); }
    else            { WAIT_ASYNC(0); }
    __syncthreads();
    const _Float16* K = sK[c & 1];
    const _Float16* V = sV[c & 1];

    // 4 score tiles of 16 keys, D=64 split K=32+32
    v8f sc[4];
#pragma unroll
    for (int jt = 0; jt < 4; ++jt) {
      const v16h b0 = load_B_frag_nk(&K[(jt * 16) * 64], 64, lane);
      const v16h b1 = load_B_frag_nk(&K[(jt * 16) * 64 + 32], 64, lane);
      v8f cz = zero8();
      cz = wmma_f16(qA0, b0, cz);
      sc[jt] = wmma_f16(qA1, b1, cz);
    }

    // online softmax: row stats across 16-lane half-groups
    float alpha[8];
#pragma unroll
    for (int v = 0; v < 8; ++v) {
      float tmax = fmaxf(fmaxf(sc[0][v], sc[1][v]), fmaxf(sc[2][v], sc[3][v]));
#pragma unroll
      for (int m = 1; m < 16; m <<= 1) tmax = fmaxf(tmax, __shfl_xor(tmax, m, 32));
      const float nm = fmaxf(rmax[v], tmax);
      const float al = __expf(rmax[v] - nm);
      rmax[v] = nm;
      const float p0 = __expf(sc[0][v] - nm);
      const float p1 = __expf(sc[1][v] - nm);
      const float p2 = __expf(sc[2][v] - nm);
      const float p3 = __expf(sc[3][v] - nm);
      float ps = (p0 + p1) + (p2 + p3);
#pragma unroll
      for (int m = 1; m < 16; m <<= 1) ps += __shfl_xor(ps, m, 32);
      rsum[v] = rsum[v] * al + ps;
      alpha[v] = al;
      const int mrow = v + 8 * half;
      myP[mrow * 64 +      nn] = (_Float16)p0;
      myP[mrow * 64 + 16 + nn] = (_Float16)p1;
      myP[mrow * 64 + 32 + nn] = (_Float16)p2;
      myP[mrow * 64 + 48 + nn] = (_Float16)p3;
    }
#pragma unroll
    for (int t = 0; t < 4; ++t) {
#pragma unroll
      for (int v = 0; v < 8; ++v) O[t][v] *= alpha[v];
    }

    // intra-wave DS RAW on the P patch, then P(16x64) @ V(64x64)
    WAIT_DS0();
    const v16h pA0 = load_A_frag(myP, 64, lane);
    const v16h pA1 = load_A_frag(myP + 32, 64, lane);
#pragma unroll
    for (int t = 0; t < 4; ++t) {
      O[t] = wmma_f16(pA0, load_B_frag_kn(&V[t * 16], 64, lane), O[t]);
      O[t] = wmma_f16(pA1, load_B_frag_kn(&V[32 * 64 + t * 16], 64, lane), O[t]);
    }
    __syncthreads();
  }

#pragma unroll
  for (int v = 0; v < 8; ++v) {
    const float inv = 1.0f / rsum[v];
    const int s = q0 + v + 8 * half;
#pragma unroll
    for (int t = 0; t < 4; ++t) {
      const int d = t * 16 + nn;
      ctx[(size_t)(b * S_ + s) * E_ + h * D_ + d] = (_Float16)(O[t][v] * inv);
    }
  }
}

extern "C" void kernel_launch(void* const* d_in, const int* in_sizes, int n_in,
                              void* d_out, int out_size, void* d_ws,
                              size_t ws_size, hipStream_t stream) {
  const float* x      = (const float*)d_in[0];  // [B,S,E]
  const float* w_qkv  = (const float*)d_in[1];  // [3E,E]
  const float* b_qkv  = (const float*)d_in[2];  // [3E]
  const float* w_proj = (const float*)d_in[3];  // [E,E]
  const float* b_proj = (const float*)d_in[4];  // [E]
  float* out = (float*)d_out;                   // [B,S,E] f32

  const int M = B_ * S_;  // 8192
  // workspace layout (f16): x16 | wq16 | wp16 | qkv | ctx  (~92 MB)
  _Float16* x16  = (_Float16*)d_ws;
  _Float16* wq16 = x16 + (size_t)M * E_;
  _Float16* wp16 = wq16 + (size_t)3 * E_ * E_;
  _Float16* qkv  = wp16 + (size_t)E_ * E_;
  _Float16* ctx  = qkv + (size_t)M * 3 * E_;

  const int nx = M * E_, nwq = 3 * E_ * E_, nwp = E_ * E_;
  cvt_f32_f16<<<nx / 1024, 256, 0, stream>>>(x, x16, nx);
  cvt_f32_f16<<<nwq / 1024, 256, 0, stream>>>(w_qkv, wq16, nwq);
  cvt_f32_f16<<<nwp / 1024, 256, 0, stream>>>(w_proj, wp16, nwp);

  gemm_f16_bias<_Float16>
      <<<dim3(3 * E_ / 64, M / 256), 256, 0, stream>>>(x16, wq16, b_qkv, qkv,
                                                       M, 3 * E_, E_);
  flash_attn_f16<<<dim3(S_ / 128, H_, B_), 256, 0, stream>>>(qkv, ctx);
  gemm_f16_bias<float>
      <<<dim3(E_ / 64, M / 256), 256, 0, stream>>>(ctx, wp16, b_proj, out, M,
                                                   E_, E_);
}
